// MultiHeadSelfAttention_79474074845704
// MI455X (gfx1250) — matrix operand order repaired and checked by reading, not compiled, not run on hardware
//
#include <hip/hip_runtime.h>

typedef __attribute__((ext_vector_type(16))) _Float16 v16h;
typedef __attribute__((ext_vector_type(8)))  _Float16 v8h;
typedef __attribute__((ext_vector_type(8)))  float    v8f;

#define NB    4
#define SEQ   1024
#define EMB   1024
#define NH    16
#define HD    64

__device__ __forceinline__ v8f wmma_f16(v16h a, v16h b, v8f c) {
  return __builtin_amdgcn_wmma_f32_16x16x32_f16(false, a, false, b, (short)0, c,
                                                false, false);
}

__device__ __forceinline__ v8f vzero8() {
  v8f z;
#pragma unroll
  for (int j = 0; j < 8; ++j) z[j] = 0.0f;
  return z;
}

// A fragment: 16x32 f16, row-major source, lda in elements.
// Lane layout (ISA 7.12.2): lanes 0-15 -> M=lane, K {0..7,16..23};
//                           lanes 16-31 -> M=lane-16, K {8..15,24..31}.
__device__ __forceinline__ v16h load_a_f16(const _Float16* src, int lda) {
  const int lane = threadIdx.x & 31;
  const int m    = lane & 15;
  const int kb   = (lane >> 4) << 3;      // 0 or 8
  const _Float16* p = src + m * lda + kb;
  v8h lo = *(const v8h*)(p);
  v8h hi = *(const v8h*)(p + 16);
  v16h a;
#pragma unroll
  for (int j = 0; j < 8; ++j) { a[j] = lo[j]; a[8 + j] = hi[j]; }
  return a;
}

// Same A fragment but converting from fp32 source.
__device__ __forceinline__ v16h load_a_f32(const float* src, int lda) {
  const int lane = threadIdx.x & 31;
  const int m    = lane & 15;
  const int kb   = (lane >> 4) << 3;
  const float* p = src + m * lda + kb;
  v16h a;
#pragma unroll
  for (int j = 0; j < 8; ++j) {
    a[j]     = (_Float16)p[j];
    a[8 + j] = (_Float16)p[16 + j];
  }
  return a;
}

// B fragment: 32x16 f16 taken from the TRANSPOSED (N x K row-major) matrix so
// each lane reads two runs of 8 contiguous halfs. Lane layout: lanes 0-15 -> N=lane,
// K {0..7,16..23}; lanes 16-31 -> N=lane-16, K {8..15,24..31}.
__device__ __forceinline__ v16h load_b_t(const _Float16* srcT, int ldt) {
  const int lane = threadIdx.x & 31;
  const int n    = lane & 15;
  const int kb   = (lane >> 4) << 3;      // 0 or 8
  const _Float16* p = srcT + n * ldt + kb;
  v8h lo = *(const v8h*)(p);
  v8h hi = *(const v8h*)(p + 16);
  v16h b;
#pragma unroll
  for (int j = 0; j < 8; ++j) { b[j] = lo[j]; b[8 + j] = hi[j]; }
  return b;
}

// Store a 16x16 f32 accumulator tile as f16, row-major dst with leading dim ldd.
__device__ __forceinline__ void store_c_f16(_Float16* dst, int ldd, v8f c) {
  const int lane = threadIdx.x & 31;
  const int n    = lane & 15;
  const int mo   = (lane >> 4) << 3;
#pragma unroll
  for (int r = 0; r < 8; ++r) dst[(mo + r) * ldd + n] = (_Float16)c[r];
}

// ---------------------------------------------------------------- kernel 0
__global__ __launch_bounds__(256) void k_cvt_w(
    const float* Wv, const float* Wk, const float* Wq, const float* Wo,
    _Float16* Wv16, _Float16* Wk16, _Float16* Wq16, _Float16* Wo16) {
  int i = blockIdx.x * 256 + threadIdx.x;
  if (i < HD * HD) {
    Wv16[i] = (_Float16)Wv[i];
    Wk16[i] = (_Float16)Wk[i];
    Wq16[i] = (_Float16)Wq[i];
  }
  if (i < EMB * EMB) Wo16[i] = (_Float16)Wo[i];
}

// ---------------------------------------------------------------- kernel 1
// Per-head projections: Q = Xq*Wq^T, K = Xk*Wk^T, V = Xv*Wv^T.
// Q,K stored [n][h][l][d]; V stored transposed [n][h][d][l].
__global__ __launch_bounds__(256) void k_qkv(
    const float* q_in, const float* k_in, const float* v_in,
    const _Float16* Wq16, const _Float16* Wk16, const _Float16* Wv16,
    _Float16* Q16, _Float16* K16, _Float16* Vt16) {
  const int wave = threadIdx.x >> 5;
  int b  = blockIdx.x;
  const int rb = b & 7;  b >>= 3;          // row block of 128
  const int t  = b % 3;  b /= 3;           // 0=Q 1=K 2=V
  const int h  = b & 15; b >>= 4;
  const int n  = b;
  const int q0 = rb * 128 + wave * 16;

  const float*    X = (t == 0) ? q_in : (t == 1) ? k_in : v_in;
  const _Float16* W = (t == 0) ? Wq16 : (t == 1) ? Wk16 : Wv16;

  const float* xrow = X + (size_t)(n * SEQ + q0) * EMB + h * HD;
  v16h a0 = load_a_f32(xrow, EMB);
  v16h a1 = load_a_f32(xrow + 32, EMB);

  v8f acc[4];
#pragma unroll
  for (int t4 = 0; t4 < 4; ++t4) {
    acc[t4] = vzero8();
    v16h b0 = load_b_t(W + (t4 * 16) * HD, HD);        // K-chunk 0..31
    v16h b1 = load_b_t(W + (t4 * 16) * HD + 32, HD);   // K-chunk 32..63
    acc[t4] = wmma_f16(a0, b0, acc[t4]);
    acc[t4] = wmma_f16(a1, b1, acc[t4]);
  }

  const int lane = threadIdx.x & 31;
  const int nl   = lane & 15;
  const int mo   = (lane >> 4) << 3;
  if (t == 2) {  // V transposed: Vt[n][h][d][l]
    _Float16* vt = Vt16 + (size_t)(n * NH + h) * HD * SEQ;
#pragma unroll
    for (int t4 = 0; t4 < 4; ++t4)
#pragma unroll
      for (int r = 0; r < 8; ++r)
        vt[(size_t)(t4 * 16 + nl) * SEQ + (q0 + mo + r)] = (_Float16)acc[t4][r];
  } else {
    _Float16* d = ((t == 0) ? Q16 : K16) +
                  ((size_t)(n * NH + h) * SEQ + q0) * HD;
#pragma unroll
    for (int t4 = 0; t4 < 4; ++t4) store_c_f16(d + t4 * 16, HD, acc[t4]);
  }
}

// ---------------------------------------------------------------- kernel 2
// Flash attention, one wave per 16 query rows of one (n,h).
__global__ __launch_bounds__(256) void k_attn(
    const _Float16* Q16, const _Float16* K16, const _Float16* Vt16,
    const int* pmask, _Float16* O16) {
  __shared__ __align__(16) _Float16 pbuf[8 * 16 * 32];  // per-wave P tile

  const int wave = threadIdx.x >> 5;
  const int lane = threadIdx.x & 31;
  const int nl   = lane & 15;
  const int mo   = (lane >> 4) << 3;

  int gw = blockIdx.x * 8 + wave;
  const int qt = gw & 63; gw >>= 6;
  const int h  = gw & 15; gw >>= 4;
  const int n  = gw;
  const int q0 = qt * 16;

  const _Float16* Qb  = Q16  + (size_t)(n * NH + h) * SEQ * HD;
  const _Float16* Kb  = K16  + (size_t)(n * NH + h) * SEQ * HD;
  const _Float16* Vtb = Vt16 + (size_t)(n * NH + h) * HD * SEQ;
  const int* pm = pmask + n * SEQ;

  v16h qa0 = load_a_f16(Qb + (size_t)q0 * HD, HD);
  v16h qa1 = load_a_f16(Qb + (size_t)q0 * HD + 32, HD);

  v8f acc[4];
#pragma unroll
  for (int t4 = 0; t4 < 4; ++t4) acc[t4] = vzero8();
  float mrow[8], lrow[8];
#pragma unroll
  for (int r = 0; r < 8; ++r) { mrow[r] = -1e30f; lrow[r] = 0.0f; }

  const float scale = 0.03125f;               // 1/sqrt(EMB) = 1/32
  const int nkb = (q0 + 16 + 31) >> 5;        // causal: key blocks of 32
  _Float16* pw = pbuf + wave * 512;

  for (int kb = 0; kb < nkb; ++kb) {
    const int k0 = kb * 32;

    // Speculative prefetch of the NEXT key block (global_prefetch_b8):
    // lanes fan out across the 32 K rows / 64 Vt rows of block kb+1 so the
    // fetch overlaps this iteration's WMMA + softmax work.
    if (kb + 1 < nkb) {
      const int kn = k0 + 32;
      __builtin_prefetch(Kb + (size_t)(kn + lane) * HD, 0, 1);
      __builtin_prefetch(Vtb + (size_t)(lane * 2) * SEQ + kn, 0, 1);
    }

    // S = Q * K^T  (two 16-key tiles)
    v8f s0 = vzero8(), s1 = vzero8();
    {
      v16h b00 = load_b_t(Kb + (size_t)(k0)*HD, HD);
      v16h b01 = load_b_t(Kb + (size_t)(k0)*HD + 32, HD);
      s0 = wmma_f16(qa0, b00, s0);
      s0 = wmma_f16(qa1, b01, s0);
      v16h b10 = load_b_t(Kb + (size_t)(k0 + 16) * HD, HD);
      v16h b11 = load_b_t(Kb + (size_t)(k0 + 16) * HD + 32, HD);
      s1 = wmma_f16(qa0, b10, s1);
      s1 = wmma_f16(qa1, b11, s1);
    }
    const int key0 = k0 + nl, key1 = k0 + 16 + nl;
    const bool pad0 = pm[key0] != 0, pad1 = pm[key1] != 0;

    float e0[8], e1[8], bm[8];
#pragma unroll
    for (int r = 0; r < 8; ++r) {
      const int q = q0 + mo + r;
      const bool ok0 = (pad0 && key0 <= q) || (key0 == q);
      const bool ok1 = (pad1 && key1 <= q) || (key1 == q);
      e0[r] = ok0 ? s0[r] * scale : -1e30f;
      e1[r] = ok1 ? s1[r] * scale : -1e30f;
      bm[r] = fmaxf(e0[r], e1[r]);
    }
    // row max across the 16-lane half (rows live replicated on 16 lanes)
#pragma unroll
    for (int r = 0; r < 8; ++r) {
      float v = bm[r];
      v = fmaxf(v, __shfl_xor(v, 1, 32));
      v = fmaxf(v, __shfl_xor(v, 2, 32));
      v = fmaxf(v, __shfl_xor(v, 4, 32));
      v = fmaxf(v, __shfl_xor(v, 8, 32));
      bm[r] = v;
    }
    float alpha[8], rs[8];
#pragma unroll
    for (int r = 0; r < 8; ++r) {
      const float mn = fmaxf(mrow[r], bm[r]);
      alpha[r] = __expf(mrow[r] - mn);
      mrow[r]  = mn;
      e0[r] = __expf(e0[r] - mn);
      e1[r] = __expf(e1[r] - mn);
      rs[r] = e0[r] + e1[r];
    }
#pragma unroll
    for (int r = 0; r < 8; ++r) {
      float v = rs[r];
      v += __shfl_xor(v, 1, 32);
      v += __shfl_xor(v, 2, 32);
      v += __shfl_xor(v, 4, 32);
      v += __shfl_xor(v, 8, 32);
      lrow[r] = lrow[r] * alpha[r] + v;
    }
#pragma unroll
    for (int t4 = 0; t4 < 4; ++t4)
#pragma unroll
      for (int r = 0; r < 8; ++r) acc[t4][r] *= alpha[r];

    // C-layout -> A-layout transpose of P via per-wave LDS tile (same-wave
    // LDS ops are processed in order, ISA 7.3).
#pragma unroll
    for (int r = 0; r < 8; ++r) {
      pw[(mo + r) * 32 + nl]      = (_Float16)e0[r];
      pw[(mo + r) * 32 + 16 + nl] = (_Float16)e1[r];
    }
    v16h pa = load_a_f16(pw, 32);

    // O += P * V  (V stored transposed: contiguous B-fragment rows)
#pragma unroll
    for (int t4 = 0; t4 < 4; ++t4) {
      v16h vb = load_b_t(Vtb + (size_t)(t4 * 16) * SEQ + k0, SEQ);
      acc[t4] = wmma_f16(pa, vb, acc[t4]);
    }
  }

  // normalize and store O at [n*SEQ+q][h*64+d]
  _Float16* ob = O16 + (size_t)(n * SEQ + q0) * EMB + h * HD;
#pragma unroll
  for (int r = 0; r < 8; ++r) lrow[r] = 1.0f / lrow[r];
#pragma unroll
  for (int t4 = 0; t4 < 4; ++t4)
#pragma unroll
    for (int r = 0; r < 8; ++r)
      ob[(size_t)(mo + r) * EMB + t4 * 16 + nl] =
          (_Float16)(acc[t4][r] * lrow[r]);
}

// ---------------------------------------------------------------- kernel 3
// out = O * Wo^T + bo  (4096x1024 x 1024x1024), fp32 output.
__global__ __launch_bounds__(256) void k_outproj(
    const _Float16* O16, const _Float16* Wo16, const float* bo, float* out) {
  const int wave = threadIdx.x >> 5;
  const int lane = threadIdx.x & 31;
  int gw = blockIdx.x * 8 + wave;
  const int ct = gw & 15; gw >>= 4;   // 16 col tiles of 64
  const int rt = gw;                  // 256 row tiles of 16
  const int r0 = rt * 16, c0 = ct * 64;

  v8f acc[4];
#pragma unroll
  for (int t4 = 0; t4 < 4; ++t4) acc[t4] = vzero8();

  const _Float16* arow = O16 + (size_t)r0 * EMB;
  for (int kc = 0; kc < EMB / 32; ++kc) {
    // Prefetch next k-chunk of the A tile (16 rows) one iteration ahead.
    if (kc + 1 < EMB / 32)
      __builtin_prefetch(arow + (size_t)(lane & 15) * EMB + (kc + 1) * 32, 0, 1);

    v16h a = load_a_f16(arow + kc * 32, EMB);
#pragma unroll
    for (int t4 = 0; t4 < 4; ++t4) {
      v16h b = load_b_t(Wo16 + (size_t)(c0 + t4 * 16) * EMB + kc * 32, EMB);
      acc[t4] = wmma_f16(a, b, acc[t4]);
    }
  }

  const int nl = lane & 15;
  const int mo = (lane >> 4) << 3;
#pragma unroll
  for (int t4 = 0; t4 < 4; ++t4) {
    const float bias = bo[c0 + t4 * 16 + nl];
#pragma unroll
    for (int r = 0; r < 8; ++r)
      out[(size_t)(r0 + mo + r) * EMB + c0 + t4 * 16 + nl] = acc[t4][r] + bias;
  }
}

// ---------------------------------------------------------------- launch
extern "C" void kernel_launch(void* const* d_in, const int* in_sizes, int n_in,
                              void* d_out, int out_size, void* d_ws,
                              size_t ws_size, hipStream_t stream) {
  (void)in_sizes; (void)n_in; (void)out_size; (void)ws_size;
  const float* values = (const float*)d_in[0];
  const float* keys   = (const float*)d_in[1];
  const float* query  = (const float*)d_in[2];
  const int*   pmask  = (const int*)d_in[3];
  // d_in[4] = causal mask: implicit (tril), not read.
  const float* Wv = (const float*)d_in[5];
  const float* Wk = (const float*)d_in[6];
  const float* Wq = (const float*)d_in[7];
  const float* Wo = (const float*)d_in[8];
  const float* bo = (const float*)d_in[9];
  float* out = (float*)d_out;

  char* ws = (char*)d_ws;
  size_t off = 0;
  _Float16* Wv16 = (_Float16*)(ws + off); off += (size_t)HD * HD * 2;
  _Float16* Wk16 = (_Float16*)(ws + off); off += (size_t)HD * HD * 2;
  _Float16* Wq16 = (_Float16*)(ws + off); off += (size_t)HD * HD * 2;
  _Float16* Wo16 = (_Float16*)(ws + off); off += (size_t)EMB * EMB * 2;
  _Float16* Q16  = (_Float16*)(ws + off); off += (size_t)NB * SEQ * EMB * 2;
  _Float16* K16  = (_Float16*)(ws + off); off += (size_t)NB * SEQ * EMB * 2;
  _Float16* Vt16 = (_Float16*)(ws + off); off += (size_t)NB * SEQ * EMB * 2;
  _Float16* O16  = (_Float16*)(ws + off); off += (size_t)NB * SEQ * EMB * 2;
  // total workspace: ~34 MB

  k_cvt_w<<<(EMB * EMB + 255) / 256, 256, 0, stream>>>(
      Wv, Wk, Wq, Wo, Wv16, Wk16, Wq16, Wo16);
  k_qkv<<<NB * NH * 3 * (SEQ / 128), 256, 0, stream>>>(
      query, keys, values, Wq16, Wk16, Wv16, Q16, K16, Vt16);
  k_attn<<<NB * NH * (SEQ / 16) / 8, 256, 0, stream>>>(
      Q16, K16, Vt16, pmask, O16);
  k_outproj<<<(NB * SEQ / 16) * (EMB / 64) / 8, 256, 0, stream>>>(
      O16, Wo16, bo, out);
}
